// CausalSelfAttention_19524921328110
// MI455X (gfx1250) — compile-verified, matrix operand order checked by reading, not run on hardware
//
#include <hip/hip_runtime.h>

// ---------------------------------------------------------------------------
// Causal self-attention, MI455X (gfx1250), bf16 WMMA pipeline.
//   B=4, T=2048, C=1024, H=16, D=64
// Stages:
//   1) convert x -> bf16 ; w_qkv,w_proj -> bf16 transposed [N][K]
//   2) QKV GEMM (WMMA bf16) -> Q[bh,t,d], K[bh,t,d], Vt[bh,d,t] (bf16)
//   3) flash attention (WMMA bf16, online softmax, wave-per-16-queries)
//   4) proj GEMM (WMMA bf16) + bias -> f32 out
// ---------------------------------------------------------------------------

#define TT   2048
#define CDIM 1024
#define HD   64
#define NH   16
#define BB   4
#define MTOT (BB * TT)        // 8192
#define N_QKV (3 * CDIM)      // 3072

typedef __attribute__((ext_vector_type(16))) __bf16 v16bf;
typedef __attribute__((ext_vector_type(8)))  float  v8f;

union FragAB { v16bf v; uint4 q[2]; };

__device__ __forceinline__ unsigned short f2bf(float f) {
  union { float f; unsigned u; } c; c.f = f;
  unsigned u = c.u;
  return (unsigned short)((u + 0x7FFFu + ((u >> 16) & 1u)) >> 16);
}

// ---------------------------------------------------------------------------
// Stage 1 kernels: dtype conversion / transpose
// ---------------------------------------------------------------------------
__global__ __launch_bounds__(256) void k_f32_to_bf16(
    const float* __restrict__ src, unsigned short* __restrict__ dst, int n) {
  int i = blockIdx.x * 256 + threadIdx.x;
  if (i < n) dst[i] = f2bf(src[i]);
}

// src is [K][N] f32; dst is [N][K] bf16 ; dst[i] with i = n*K + k
__global__ __launch_bounds__(256) void k_transpose_to_bf16(
    const float* __restrict__ src, unsigned short* __restrict__ dst,
    int K, int N) {
  int i = blockIdx.x * 256 + threadIdx.x;
  if (i >= K * N) return;
  int n = i / K, k = i - n * K;
  dst[i] = f2bf(src[k * N + n]);
}

// ---------------------------------------------------------------------------
// Stage 2: QKV GEMM.  C[8192,3072] = Xb[8192,1024] @ Wqkv + bias, scattered to
// Q[bh,t,d], K[bh,t,d], Vt[bh,d,t] in bf16.
// Workgroup: 256 threads (8 waves), 128x128 output tile, k-step 32.
// Wave (wm,wn) owns a 32x64 sub-tile = 2x4 WMMA accumulators.
// ---------------------------------------------------------------------------
__global__ __launch_bounds__(256) void k_qkv_gemm(
    const unsigned short* __restrict__ Xb,
    const unsigned short* __restrict__ WT,   // [3072][1024] bf16
    const float* __restrict__ bias,
    unsigned short* __restrict__ Qo,
    unsigned short* __restrict__ Ko,
    unsigned short* __restrict__ Vto) {
  __shared__ __align__(16) unsigned short As[128 * 32];
  __shared__ __align__(16) unsigned short Bs[128 * 32];

  const int tid  = threadIdx.x;
  const int lane = tid & 31, wave = tid >> 5;
  const int half = lane >> 4, r15 = lane & 15;
  const int wm = wave & 3, wn = wave >> 2;
  const int M0 = blockIdx.y * 128, N0 = blockIdx.x * 128;

  const v8f vz = {0.f, 0.f, 0.f, 0.f, 0.f, 0.f, 0.f, 0.f};
  v8f acc[2][4];
#pragma unroll
  for (int i = 0; i < 2; ++i)
#pragma unroll
    for (int j = 0; j < 4; ++j) acc[i][j] = vz;

  for (int k0 = 0; k0 < CDIM; k0 += 32) {
#pragma unroll
    for (int t = 0; t < 2; ++t) {           // 512 16B chunks per 8KB tile
      int c = tid + t * 256;
      int row = c >> 2, cc = c & 3;
      *(uint4*)&As[row * 32 + cc * 8] =
          *(const uint4*)&Xb[(size_t)(M0 + row) * CDIM + k0 + cc * 8];
      *(uint4*)&Bs[row * 32 + cc * 8] =
          *(const uint4*)&WT[(size_t)(N0 + row) * CDIM + k0 + cc * 8];
    }
    if (k0 + 32 < CDIM) {
      __builtin_prefetch(&Xb[(size_t)(M0 + (tid >> 2)) * CDIM + k0 + 32], 0, 1);
      __builtin_prefetch(&WT[(size_t)(N0 + (tid >> 2)) * CDIM + k0 + 32], 0, 1);
    }
    __syncthreads();

    FragAB a[2], b[4];
#pragma unroll
    for (int i = 0; i < 2; ++i) {
      int row = wm * 32 + i * 16 + r15;
      a[i].q[0] = *(const uint4*)&As[row * 32 + half * 8];
      a[i].q[1] = *(const uint4*)&As[row * 32 + 16 + half * 8];
    }
#pragma unroll
    for (int j = 0; j < 4; ++j) {
      int row = wn * 64 + j * 16 + r15;
      b[j].q[0] = *(const uint4*)&Bs[row * 32 + half * 8];
      b[j].q[1] = *(const uint4*)&Bs[row * 32 + 16 + half * 8];
    }
#pragma unroll
    for (int i = 0; i < 2; ++i)
#pragma unroll
      for (int j = 0; j < 4; ++j)
        acc[i][j] = __builtin_amdgcn_wmma_f32_16x16x32_bf16(
            false, a[i].v, false, b[j].v, (short)0, acc[i][j], false, false);
    __syncthreads();
  }

  // Epilogue: C/D layout -> element (M = r + 8*half, N = r15) per VGPR r.
#pragma unroll
  for (int i = 0; i < 2; ++i)
#pragma unroll
    for (int j = 0; j < 4; ++j)
#pragma unroll
      for (int r = 0; r < 8; ++r) {
        int row = M0 + wm * 32 + i * 16 + r + 8 * half;
        int col = N0 + wn * 64 + j * 16 + r15;
        float v = acc[i][j][r] + bias[col];
        int s = col >> 10, rem = col & 1023;
        int h = rem >> 6, d = rem & 63;
        int bI = row >> 11, t = row & 2047;
        int bh = bI * NH + h;
        unsigned short bv = f2bf(v);
        if (s == 0)      Qo[((size_t)bh * TT + t) * HD + d] = bv;
        else if (s == 1) Ko[((size_t)bh * TT + t) * HD + d] = bv;
        else             Vto[((size_t)bh * HD + d) * TT + t] = bv;
      }
}

// ---------------------------------------------------------------------------
// Stage 3: flash attention.  One wave = 16 query rows; 8 waves per block.
// S = Q Kt (2 col-tiles x 2 k-steps), online softmax, O += P V (4 WMMAs).
// K fragments: direct b128 loads from K[bh,t,d] (row = key, contiguous d).
// V fragments: direct b128 loads from Vt[bh,d,t] (row = d, contiguous keys).
// P re-layout C->A fragment via 1KB per-wave LDS scratch.
// ---------------------------------------------------------------------------
__global__ __launch_bounds__(256) void k_attn(
    const unsigned short* __restrict__ Q,
    const unsigned short* __restrict__ K,
    const unsigned short* __restrict__ Vt,
    unsigned short* __restrict__ Yb) {
  __shared__ __align__(16) unsigned short Ps[8][512];

  const int tid  = threadIdx.x;
  const int lane = tid & 31, wave = tid >> 5;
  const int half = lane >> 4, r15 = lane & 15;
  const int wid = blockIdx.x * 8 + wave;       // 0..8191
  const int bh  = wid >> 7;                    // 0..63
  const int q0  = (wid & 127) << 4;            // query tile base

  const unsigned short* Qb = Q  + (size_t)bh * TT * HD;
  const unsigned short* Kb = K  + (size_t)bh * TT * HD;
  const unsigned short* Vb = Vt + (size_t)bh * HD * TT;
  unsigned short* myP = &Ps[wave][0];

  // Q fragments (persist in registers): A 16x32 per d-halves
  FragAB qf[2];
#pragma unroll
  for (int s = 0; s < 2; ++s) {
    size_t base = (size_t)(q0 + r15) * HD + s * 32 + half * 8;
    qf[s].q[0] = *(const uint4*)&Qb[base];
    qf[s].q[1] = *(const uint4*)&Qb[base + 16];
  }

  const v8f vz = {0.f, 0.f, 0.f, 0.f, 0.f, 0.f, 0.f, 0.f};
  v8f o[4];
#pragma unroll
  for (int t = 0; t < 4; ++t) o[t] = vz;
  float m_[8], l_[8];
#pragma unroll
  for (int r = 0; r < 8; ++r) { m_[r] = -1e30f; l_[r] = 0.f; }

  const float scale = 0.125f;                  // 1/sqrt(64)
  const int nkb = ((q0 + 15) >> 5) + 1;        // causal: key blocks of 32

  for (int jb0 = 0; jb0 < nkb; ++jb0) {
    const int jb = jb0 * 32;

    // S = Q @ K^T : two 16x16 column tiles, each accumulated over 2 k-steps
    v8f s[2];
#pragma unroll
    for (int c = 0; c < 2; ++c) {
      v8f sc = vz;
#pragma unroll
      for (int ks = 0; ks < 2; ++ks) {
        FragAB bf;
        size_t base = (size_t)(jb + c * 16 + r15) * HD + ks * 32 + half * 8;
        bf.q[0] = *(const uint4*)&Kb[base];
        bf.q[1] = *(const uint4*)&Kb[base + 16];
        sc = __builtin_amdgcn_wmma_f32_16x16x32_bf16(
            false, qf[ks].v, false, bf.v, (short)0, sc, false, false);
      }
      s[c] = sc;
    }

    // scale + causal mask (element row = q0 + r + 8*half, col = jb+c*16+r15)
#pragma unroll
    for (int c = 0; c < 2; ++c)
#pragma unroll
      for (int r = 0; r < 8; ++r) {
        int key = jb + c * 16 + r15;
        int qq  = q0 + r + 8 * half;
        float v = s[c][r] * scale;
        s[c][r] = (key > qq) ? -1e30f : v;
      }

    // row max across 16 lanes of each half-wave
    float mb[8];
#pragma unroll
    for (int r = 0; r < 8; ++r) mb[r] = fmaxf(s[0][r], s[1][r]);
#pragma unroll
    for (int off = 1; off < 16; off <<= 1)
#pragma unroll
      for (int r = 0; r < 8; ++r)
        mb[r] = fmaxf(mb[r], __shfl_xor(mb[r], off, 32));

    float alpha[8];
#pragma unroll
    for (int r = 0; r < 8; ++r) {
      float mn = fmaxf(m_[r], mb[r]);
      alpha[r] = __expf(m_[r] - mn);
      m_[r] = mn;
    }

    // P = exp(S - m), row sum
    float rs[8];
#pragma unroll
    for (int r = 0; r < 8; ++r) {
      float p0 = __expf(s[0][r] - m_[r]);
      float p1 = __expf(s[1][r] - m_[r]);
      s[0][r] = p0; s[1][r] = p1;
      rs[r] = p0 + p1;
    }
#pragma unroll
    for (int off = 1; off < 16; off <<= 1)
#pragma unroll
      for (int r = 0; r < 8; ++r) rs[r] += __shfl_xor(rs[r], off, 32);
#pragma unroll
    for (int r = 0; r < 8; ++r) l_[r] = l_[r] * alpha[r] + rs[r];

    // rescale running O
#pragma unroll
    for (int t = 0; t < 4; ++t)
#pragma unroll
      for (int r = 0; r < 8; ++r) o[t][r] *= alpha[r];

    // P (C layout) -> A fragment via per-wave LDS scratch [16 rows][32 keys]
#pragma unroll
    for (int c = 0; c < 2; ++c)
#pragma unroll
      for (int r = 0; r < 8; ++r)
        myP[(r + 8 * half) * 32 + c * 16 + r15] = f2bf(s[c][r]);
    asm volatile("s_wait_dscnt 0" ::: "memory");

    FragAB pa;
    pa.q[0] = *(const uint4*)&myP[r15 * 32 + half * 8];
    pa.q[1] = *(const uint4*)&myP[r15 * 32 + 16 + half * 8];

    // O += P @ V : 4 d-column tiles, B from Vt (keys contiguous)
#pragma unroll
    for (int t = 0; t < 4; ++t) {
      FragAB bv;
      size_t base = (size_t)(t * 16 + r15) * TT + jb + half * 8;
      bv.q[0] = *(const uint4*)&Vb[base];
      bv.q[1] = *(const uint4*)&Vb[base + 16];
      o[t] = __builtin_amdgcn_wmma_f32_16x16x32_bf16(
          false, pa.v, false, bv.v, (short)0, o[t], false, false);
    }
  }

  // epilogue: y[b, t, h*64 + d] bf16
  const int bI = bh >> 4, h = bh & 15;
#pragma unroll
  for (int r = 0; r < 8; ++r) {
    float inv = 1.0f / l_[r];
    int t = q0 + r + 8 * half;
    size_t rowbase = ((size_t)bI * TT + t) * CDIM + h * HD;
#pragma unroll
    for (int tt = 0; tt < 4; ++tt)
      Yb[rowbase + tt * 16 + r15] = f2bf(o[tt][r] * inv);
  }
}

// ---------------------------------------------------------------------------
// Stage 4: output projection.  out[8192,1024] = Yb @ Wproj + bias (f32 out)
// ---------------------------------------------------------------------------
__global__ __launch_bounds__(256) void k_proj_gemm(
    const unsigned short* __restrict__ Yb,
    const unsigned short* __restrict__ WT,   // [1024][1024] bf16
    const float* __restrict__ bias,
    float* __restrict__ out) {
  __shared__ __align__(16) unsigned short As[128 * 32];
  __shared__ __align__(16) unsigned short Bs[128 * 32];

  const int tid  = threadIdx.x;
  const int lane = tid & 31, wave = tid >> 5;
  const int half = lane >> 4, r15 = lane & 15;
  const int wm = wave & 3, wn = wave >> 2;
  const int M0 = blockIdx.y * 128, N0 = blockIdx.x * 128;

  const v8f vz = {0.f, 0.f, 0.f, 0.f, 0.f, 0.f, 0.f, 0.f};
  v8f acc[2][4];
#pragma unroll
  for (int i = 0; i < 2; ++i)
#pragma unroll
    for (int j = 0; j < 4; ++j) acc[i][j] = vz;

  for (int k0 = 0; k0 < CDIM; k0 += 32) {
#pragma unroll
    for (int t = 0; t < 2; ++t) {
      int c = tid + t * 256;
      int row = c >> 2, cc = c & 3;
      *(uint4*)&As[row * 32 + cc * 8] =
          *(const uint4*)&Yb[(size_t)(M0 + row) * CDIM + k0 + cc * 8];
      *(uint4*)&Bs[row * 32 + cc * 8] =
          *(const uint4*)&WT[(size_t)(N0 + row) * CDIM + k0 + cc * 8];
    }
    __syncthreads();

    FragAB a[2], b[4];
#pragma unroll
    for (int i = 0; i < 2; ++i) {
      int row = wm * 32 + i * 16 + r15;
      a[i].q[0] = *(const uint4*)&As[row * 32 + half * 8];
      a[i].q[1] = *(const uint4*)&As[row * 32 + 16 + half * 8];
    }
#pragma unroll
    for (int j = 0; j < 4; ++j) {
      int row = wn * 64 + j * 16 + r15;
      b[j].q[0] = *(const uint4*)&Bs[row * 32 + half * 8];
      b[j].q[1] = *(const uint4*)&Bs[row * 32 + 16 + half * 8];
    }
#pragma unroll
    for (int i = 0; i < 2; ++i)
#pragma unroll
      for (int j = 0; j < 4; ++j)
        acc[i][j] = __builtin_amdgcn_wmma_f32_16x16x32_bf16(
            false, a[i].v, false, b[j].v, (short)0, acc[i][j], false, false);
    __syncthreads();
  }

#pragma unroll
  for (int i = 0; i < 2; ++i)
#pragma unroll
    for (int j = 0; j < 4; ++j)
#pragma unroll
      for (int r = 0; r < 8; ++r) {
        int row = M0 + wm * 32 + i * 16 + r + 8 * half;
        int col = N0 + wn * 64 + j * 16 + r15;
        out[(size_t)row * CDIM + col] = acc[i][j][r] + bias[col];
      }
}

// ---------------------------------------------------------------------------
// Host launcher
// ---------------------------------------------------------------------------
extern "C" void kernel_launch(void* const* d_in, const int* in_sizes, int n_in,
                              void* d_out, int out_size, void* d_ws,
                              size_t ws_size, hipStream_t stream) {
  const float* x      = (const float*)d_in[0];
  const float* w_qkv  = (const float*)d_in[1];
  const float* b_qkv  = (const float*)d_in[2];
  const float* w_proj = (const float*)d_in[3];
  const float* b_proj = (const float*)d_in[4];
  float* out = (float*)d_out;

  char* ws = (char*)d_ws;
  size_t off = 0;
  unsigned short* xb     = (unsigned short*)(ws + off); off += (size_t)MTOT * CDIM * 2;   // 16 MB
  unsigned short* wqkvT  = (unsigned short*)(ws + off); off += (size_t)N_QKV * CDIM * 2;  //  6 MB
  unsigned short* wprojT = (unsigned short*)(ws + off); off += (size_t)CDIM * CDIM * 2;   //  2 MB
  unsigned short* Qb     = (unsigned short*)(ws + off); off += (size_t)MTOT * CDIM * 2;   // 16 MB
  unsigned short* Kb     = (unsigned short*)(ws + off); off += (size_t)MTOT * CDIM * 2;   // 16 MB
  unsigned short* Vtb    = (unsigned short*)(ws + off); off += (size_t)MTOT * CDIM * 2;   // 16 MB
  unsigned short* Yb     = (unsigned short*)(ws + off); off += (size_t)MTOT * CDIM * 2;   // 16 MB

  // Stage 1: conversions
  k_f32_to_bf16<<<(MTOT * CDIM) / 256, 256, 0, stream>>>(x, xb, MTOT * CDIM);
  k_transpose_to_bf16<<<(N_QKV * CDIM) / 256, 256, 0, stream>>>(w_qkv, wqkvT, CDIM, N_QKV);
  k_transpose_to_bf16<<<(CDIM * CDIM) / 256, 256, 0, stream>>>(w_proj, wprojT, CDIM, CDIM);

  // Stage 2: QKV GEMM -> Q, K, Vt (bf16)
  k_qkv_gemm<<<dim3(N_QKV / 128, MTOT / 128), 256, 0, stream>>>(
      xb, wqkvT, b_qkv, Qb, Kb, Vtb);

  // Stage 3: flash attention -> Yb (bf16)
  k_attn<<<(BB * NH * (TT / 16)) / 8, 256, 0, stream>>>(Qb, Kb, Vtb, Yb);

  // Stage 4: projection -> out (f32)
  k_proj_gemm<<<dim3(CDIM / 128, MTOT / 128), 256, 0, stream>>>(
      Yb, wprojT, b_proj, out);
}
